// siamese_fc_cos_assemble_5059471475118
// MI455X (gfx1250) — matrix operand order repaired and checked by reading, not compiled, not run on hardware
//
#include <hip/hip_runtime.h>
#include <math.h>

typedef __attribute__((ext_vector_type(16))) _Float16 v16h;
typedef __attribute__((ext_vector_type(8)))  _Float16 v8h;
typedef __attribute__((ext_vector_type(8)))  float    v8f;

#define NROW 65536           // N pairs
#define BTOT 131072          // 2N rows through the MLP
#define DIN  189
#define DPAD 192
#define H2   32
#define EPSB 1e-5f

// ---------------- K1: column sums / sumsq of raw concat(target, candidate) ----
__global__ __launch_bounds__(256) void bn0_stats_k(const float* __restrict__ tgt,
                                                   const float* __restrict__ cnd,
                                                   float* __restrict__ stats0) {
  const int tid = threadIdx.x;
  const int c0 = tid, c1 = tid + 256, c2 = tid + 512;
  const bool has1 = (c1 < 756), has2 = (c2 < 756);
  float s0 = 0.f, q0 = 0.f, s1 = 0.f, q1 = 0.f, s2 = 0.f, q2 = 0.f;
  const int r0 = blockIdx.x * 512;
  for (int r = 0; r < 512; ++r) {
    const int b = r0 + r;
    const float* row = (b < NROW) ? (tgt + (size_t)b * 756)
                                  : (cnd + (size_t)(b - NROW) * 756);
    { float v = row[c0]; s0 += v; q0 += v * v; }
    if (has1) { float v = row[c1]; s1 += v; q1 += v * v; }
    if (has2) { float v = row[c2]; s2 += v; q2 += v * v; }
  }
  __hip_atomic_fetch_add(&stats0[c0],       s0, __ATOMIC_RELAXED, __HIP_MEMORY_SCOPE_AGENT);
  __hip_atomic_fetch_add(&stats0[756 + c0], q0, __ATOMIC_RELAXED, __HIP_MEMORY_SCOPE_AGENT);
  if (has1) {
    __hip_atomic_fetch_add(&stats0[c1],       s1, __ATOMIC_RELAXED, __HIP_MEMORY_SCOPE_AGENT);
    __hip_atomic_fetch_add(&stats0[756 + c1], q1, __ATOMIC_RELAXED, __HIP_MEMORY_SCOPE_AGENT);
  }
  if (has2) {
    __hip_atomic_fetch_add(&stats0[c2],       s2, __ATOMIC_RELAXED, __HIP_MEMORY_SCOPE_AGENT);
    __hip_atomic_fetch_add(&stats0[756 + c2], q2, __ATOMIC_RELAXED, __HIP_MEMORY_SCOPE_AGENT);
  }
}

// ---------------- K2a: BN0 fold constants: s0 = g*rsqrt(var+eps), c0 = b - m*s0
__global__ void bn0_fold_k(const float* __restrict__ stats0, const float* __restrict__ g0,
                           const float* __restrict__ b0v, float* __restrict__ s0c0) {
  int i = blockIdx.x * 256 + threadIdx.x;
  if (i < 756) {
    float mean = stats0[i] * (1.f / BTOT);
    float var  = stats0[756 + i] * (1.f / BTOT) - mean * mean;
    float s = g0[i] * rsqrtf(var + EPSB);
    s0c0[i] = s;
    s0c0[756 + i] = b0v[i] - mean * s;
  }
}

// ---------------- K2b: fold BN0 scale into W1, convert W1/W2 -> padded f16 ----
__global__ void fold_w_k(const float* __restrict__ W1, const float* __restrict__ W2,
                         const float* __restrict__ s0c0,
                         _Float16* __restrict__ W1h, _Float16* __restrict__ W2h) {
  int idx = blockIdx.x * 256 + threadIdx.x;
  const int n1 = 4 * DPAD * DPAD;
  if (idx < n1) {
    int d = idx / (DPAD * DPAD);
    int rem = idx % (DPAD * DPAD);
    int o = rem / DPAD, k = rem % DPAD;
    float w = 0.f;
    if (o < DIN && k < DIN)
      w = W1[(size_t)(d * DIN + o) * 756 + d * DIN + k] * s0c0[d * DIN + k];
    W1h[idx] = (_Float16)w;
  } else {
    int j = idx - n1;
    if (j < 4 * H2 * DPAD) {
      int d = j / (H2 * DPAD);
      int rem = j % (H2 * DPAD);
      int o = rem / DPAD, k = rem % DPAD;
      float w = 0.f;
      if (k < DIN)
        w = W2[(size_t)(d * H2 + o) * 756 + d * DIN + k];
      W2h[j] = (_Float16)w;
    }
  }
}

// ---------------- K2c: bias1[o] = sum_k c0[k] * W1[o,k] (per detector block) --
__global__ void bias1_k(const float* __restrict__ W1, const float* __restrict__ s0c0,
                        float* __restrict__ bias1) {
  int idx = blockIdx.x * 256 + threadIdx.x;
  if (idx < 4 * DPAD) {
    int d = idx / DPAD, o = idx % DPAD;
    float s = 0.f;
    if (o < DIN) {
      const float* wrow = W1 + (size_t)(d * DIN + o) * 756 + d * DIN;
      const float* c0 = s0c0 + 756 + d * DIN;
      for (int k = 0; k < DIN; ++k) s += c0[k] * wrow[k];
    }
    bias1[idx] = s;
  }
}

// ---------------- K3: GEMM1  h1[b, d*192+n] = X_f16 @ W1h^T + bias1  (WMMA) ---
// 256 threads = 8 waves; 128-row batch chunk per WG.
// wave = (mrow group 0..3) x (N-half 0..1): each wave computes 2 M-tiles x 6
// N-tiles, so every B fragment is reused across 2 WMMAs (and A across 6).
__global__ __launch_bounds__(256) void gemm1_k(const float* __restrict__ tgt,
                                               const float* __restrict__ cnd,
                                               const _Float16* __restrict__ W1h,
                                               const float* __restrict__ bias1,
                                               _Float16* __restrict__ h1,
                                               float* __restrict__ stats1) {
  const int d = blockIdx.y;
  const int b0 = blockIdx.x * 128;
  __shared__ __align__(16) _Float16 Xs[128][DPAD];   // 48 KB f16 staging
  __shared__ float csum[DPAD], csq[DPAD];
  const int tid = threadIdx.x;
  for (int i = tid; i < DPAD; i += 256) { csum[i] = 0.f; csq[i] = 0.f; }
  for (int idx = tid; idx < 128 * DPAD; idx += 256) {
    int r = idx / DPAD, c = idx % DPAD;
    int b = b0 + r;
    float v = 0.f;
    if (c < DIN)
      v = (b < NROW) ? tgt[(size_t)b * 756 + d * DIN + c]
                     : cnd[(size_t)(b - NROW) * 756 + d * DIN + c];
    Xs[r][c] = (_Float16)v;
  }
  __syncthreads();

  const int wave = tid >> 5, lane = tid & 31;
  const int lm = lane & 15, hi = lane >> 4;
  const int mrow = (wave & 3) * 32;       // 2 M-tiles: rows mrow..mrow+31
  const int nbase = (wave >> 2) * 6;      // 6 N-tiles: nbase..nbase+5

  v8f acc[2][6];
#pragma unroll
  for (int nt = 0; nt < 6; ++nt) {
    float bv = bias1[d * DPAD + (nbase + nt) * 16 + lm];
    v8f init = (v8f){bv, bv, bv, bv, bv, bv, bv, bv};
    acc[0][nt] = init;
    acc[1][nt] = init;
  }

  for (int kc = 0; kc < 6; ++kc) {
    const int ka = kc * 32 + hi * 16;
    v16h a0, a1;
    {
      v8h lo = *(const v8h*)&Xs[mrow + lm][ka];
      v8h hh = *(const v8h*)&Xs[mrow + lm][ka + 8];
#pragma unroll
      for (int i = 0; i < 8; ++i) { a0[i] = lo[i]; a0[8 + i] = hh[i]; }
    }
    {
      v8h lo = *(const v8h*)&Xs[mrow + 16 + lm][ka];
      v8h hh = *(const v8h*)&Xs[mrow + 16 + lm][ka + 8];
#pragma unroll
      for (int i = 0; i < 8; ++i) { a1[i] = lo[i]; a1[8 + i] = hh[i]; }
    }
    const int kb = kc * 32 + hi * 16;
#pragma unroll
    for (int nt = 0; nt < 6; ++nt) {
      const v16h bfrag =
          *(const v16h*)(W1h + ((size_t)(d * DPAD + (nbase + nt) * 16 + lm)) * DPAD + kb);
      acc[0][nt] = __builtin_amdgcn_wmma_f32_16x16x32_f16(false, a0, false, bfrag,
                                                          (short)0, acc[0][nt], false, false);
      acc[1][nt] = __builtin_amdgcn_wmma_f32_16x16x32_f16(false, a1, false, bfrag,
                                                          (short)0, acc[1][nt], false, false);
    }
  }

#pragma unroll
  for (int mt = 0; mt < 2; ++mt) {
#pragma unroll
    for (int nt = 0; nt < 6; ++nt) {
      const int n = (nbase + nt) * 16 + lm;
      float s = 0.f, sq = 0.f;
#pragma unroll
      for (int r = 0; r < 8; ++r) {
        float v = acc[mt][nt][r];
        int row = b0 + mrow + mt * 16 + r + hi * 8;
        h1[(size_t)row * 768 + d * DPAD + n] = (_Float16)v;
        s += v; sq += v * v;
      }
      if (n < DIN) {
        __hip_atomic_fetch_add(&csum[n], s,  __ATOMIC_RELAXED, __HIP_MEMORY_SCOPE_WORKGROUP);
        __hip_atomic_fetch_add(&csq[n],  sq, __ATOMIC_RELAXED, __HIP_MEMORY_SCOPE_WORKGROUP);
      }
    }
  }
  __syncthreads();
  for (int c = tid; c < DIN; c += 256) {
    __hip_atomic_fetch_add(&stats1[d * DPAD + c],       csum[c], __ATOMIC_RELAXED, __HIP_MEMORY_SCOPE_AGENT);
    __hip_atomic_fetch_add(&stats1[768 + d * DPAD + c], csq[c],  __ATOMIC_RELAXED, __HIP_MEMORY_SCOPE_AGENT);
  }
}

// ---------------- K4: BN1 fold constants ------------------------------------
__global__ void bn1_fold_k(const float* __restrict__ stats1, const float* __restrict__ g1,
                           const float* __restrict__ b1, float* __restrict__ s1c1) {
  int idx = blockIdx.x * 256 + threadIdx.x;
  if (idx < 768) {
    int d = idx / DPAD, n = idx % DPAD;
    float s = 0.f, c = 0.f;
    if (n < DIN) {
      float mean = stats1[idx] * (1.f / BTOT);
      float var  = stats1[768 + idx] * (1.f / BTOT) - mean * mean;
      s = g1[d * DIN + n] * rsqrtf(var + EPSB);
      c = b1[d * DIN + n] - mean * s;
    }
    s1c1[idx] = s;
    s1c1[768 + idx] = c;
  }
}

// ---------------- K5: GEMM2  h2 = sigmoid(BN1(h1)) @ W2h^T  (WMMA) -----------
__global__ __launch_bounds__(128) void gemm2_k(const _Float16* __restrict__ h1,
                                               const float* __restrict__ s1c1,
                                               const _Float16* __restrict__ W2h,
                                               float* __restrict__ h2,
                                               float* __restrict__ stats2) {
  const int d = blockIdx.y;
  const int b0 = blockIdx.x * 64;
  __shared__ __align__(16) _Float16 As[64][DPAD];
  __shared__ float csum[H2], csq[H2];
  const int tid = threadIdx.x;
  if (tid < H2) { csum[tid] = 0.f; csq[tid] = 0.f; }
  for (int idx = tid; idx < 64 * DPAD; idx += 128) {
    int r = idx / DPAD, c = idx % DPAD;
    float hv = (float)h1[(size_t)(b0 + r) * 768 + d * DPAD + c];
    float a = hv * s1c1[d * DPAD + c] + s1c1[768 + d * DPAD + c];
    a = 1.f / (1.f + __expf(-a));
    As[r][c] = (_Float16)a;
  }
  __syncthreads();

  const int wave = tid >> 5, lane = tid & 31;
  const int lm = lane & 15, hi = lane >> 4;
  const int m = wave * 16 + lm;

  v8f acc[2];
  acc[0] = (v8f){0.f, 0.f, 0.f, 0.f, 0.f, 0.f, 0.f, 0.f};
  acc[1] = acc[0];

  for (int kc = 0; kc < 6; ++kc) {
    const int ka = kc * 32 + hi * 16;
    v8h alo = *(const v8h*)&As[m][ka];
    v8h ahi = *(const v8h*)&As[m][ka + 8];
    v16h a;
#pragma unroll
    for (int i = 0; i < 8; ++i) { a[i] = alo[i]; a[8 + i] = ahi[i]; }
    const int kb = kc * 32 + hi * 16;
#pragma unroll
    for (int nt = 0; nt < 2; ++nt) {
      const v16h bfrag =
          *(const v16h*)(W2h + ((size_t)(d * H2 + nt * 16 + lm)) * DPAD + kb);
      acc[nt] = __builtin_amdgcn_wmma_f32_16x16x32_f16(false, a, false, bfrag,
                                                       (short)0, acc[nt], false, false);
    }
  }

#pragma unroll
  for (int nt = 0; nt < 2; ++nt) {
    const int n = nt * 16 + lm;
    const int col = d * H2 + n;
    float s = 0.f, sq = 0.f;
#pragma unroll
    for (int r = 0; r < 8; ++r) {
      float v = acc[nt][r];
      int row = b0 + wave * 16 + r + hi * 8;
      h2[(size_t)row * 128 + col] = v;
      s += v; sq += v * v;
    }
    __hip_atomic_fetch_add(&csum[n], s,  __ATOMIC_RELAXED, __HIP_MEMORY_SCOPE_WORKGROUP);
    __hip_atomic_fetch_add(&csq[n],  sq, __ATOMIC_RELAXED, __HIP_MEMORY_SCOPE_WORKGROUP);
  }
  __syncthreads();
  if (tid < H2) {
    __hip_atomic_fetch_add(&stats2[d * H2 + tid],       csum[tid], __ATOMIC_RELAXED, __HIP_MEMORY_SCOPE_AGENT);
    __hip_atomic_fetch_add(&stats2[128 + d * H2 + tid], csq[tid],  __ATOMIC_RELAXED, __HIP_MEMORY_SCOPE_AGENT);
  }
}

// ---------------- K6: BN2 fold constants ------------------------------------
__global__ void bn2_fold_k(const float* __restrict__ stats2, const float* __restrict__ g2,
                           const float* __restrict__ b2, float* __restrict__ s2c2) {
  int i = threadIdx.x;  // 128
  float mean = stats2[i] * (1.f / BTOT);
  float var  = stats2[128 + i] * (1.f / BTOT) - mean * mean;
  float s = g2[i] * rsqrtf(var + EPSB);
  s2c2[i] = s;
  s2c2[128 + i] = b2[i] - mean * s;
}

// ---------------- K7: feature = sigmoid(BN2(h2)); cosine sim per (pair, det) -
__global__ __launch_bounds__(256) void final_k(const float* __restrict__ h2,
                                               const float* __restrict__ s2c2,
                                               float* __restrict__ out) {
  int t = blockIdx.x * 256 + threadIdx.x;   // t < N*4
  int d = t & 3, j = t >> 2;
  const float* ht = h2 + (size_t)j * 128 + d * H2;
  const float* hc = h2 + (size_t)(j + NROW) * 128 + d * H2;
  float* feat = out + (size_t)NROW * 4;
  float dot = 0.f, n2t = 0.f, n2c = 0.f;
#pragma unroll
  for (int i = 0; i < H2; ++i) {
    int col = d * H2 + i;
    float s = s2c2[col], c = s2c2[128 + col];
    float tv = 1.f / (1.f + __expf(-(ht[i] * s + c)));
    float cv = 1.f / (1.f + __expf(-(hc[i] * s + c)));
    feat[(size_t)j * 128 + col] = tv;
    feat[(size_t)(j + NROW) * 128 + col] = cv;
    dot += tv * cv;
    n2t += tv * tv;
    n2c += cv * cv;
  }
  float den = 1e-8f + sqrtf(n2t) * sqrtf(n2c);
  float sim = fminf(fabsf(dot / den), 1.f);
  out[(size_t)j * 4 + d] = sim;
}

// ---------------------------------------------------------------------------
extern "C" void kernel_launch(void* const* d_in, const int* in_sizes, int n_in,
                              void* d_out, int out_size, void* d_ws, size_t ws_size,
                              hipStream_t stream) {
  (void)in_sizes; (void)n_in; (void)out_size; (void)ws_size;
  const float* cand = (const float*)d_in[0];
  const float* tgt  = (const float*)d_in[1];
  const float* g0   = (const float*)d_in[2];
  const float* b0v  = (const float*)d_in[3];
  const float* W1   = (const float*)d_in[4];
  const float* g1   = (const float*)d_in[5];
  const float* b1   = (const float*)d_in[6];
  const float* W2   = (const float*)d_in[7];
  const float* g2   = (const float*)d_in[8];
  const float* b2   = (const float*)d_in[9];
  float* out = (float*)d_out;

  char* ws = (char*)d_ws;
  size_t off = 0;
  auto alloc = [&](size_t bytes) -> void* {
    void* p = ws + off;
    off = (off + bytes + 255) & ~(size_t)255;
    return p;
  };
  float*    stats0 = (float*)alloc(1512 * sizeof(float));
  float*    s0c0   = (float*)alloc(1512 * sizeof(float));
  _Float16* W1h    = (_Float16*)alloc((size_t)4 * DPAD * DPAD * sizeof(_Float16));
  float*    bias1  = (float*)alloc(768 * sizeof(float));
  _Float16* W2h    = (_Float16*)alloc((size_t)4 * H2 * DPAD * sizeof(_Float16));
  float*    stats1 = (float*)alloc(1536 * sizeof(float));
  float*    s1c1   = (float*)alloc(1536 * sizeof(float));
  float*    stats2 = (float*)alloc(256 * sizeof(float));
  float*    s2c2   = (float*)alloc(256 * sizeof(float));
  _Float16* h1     = (_Float16*)alloc((size_t)BTOT * 768 * sizeof(_Float16));
  float*    h2     = (float*)alloc((size_t)BTOT * 128 * sizeof(float));

  hipMemsetAsync(stats0, 0, 1512 * sizeof(float), stream);
  hipMemsetAsync(stats1, 0, 1536 * sizeof(float), stream);
  hipMemsetAsync(stats2, 0,  256 * sizeof(float), stream);

  bn0_stats_k<<<256, 256, 0, stream>>>(tgt, cand, stats0);
  bn0_fold_k<<<3, 256, 0, stream>>>(stats0, g0, b0v, s0c0);
  fold_w_k<<<672, 256, 0, stream>>>(W1, W2, s0c0, W1h, W2h);
  bias1_k<<<3, 256, 0, stream>>>(W1, s0c0, bias1);

  gemm1_k<<<dim3(BTOT / 128, 4), 256, 0, stream>>>(tgt, cand, W1h, bias1, h1, stats1);
  bn1_fold_k<<<3, 256, 0, stream>>>(stats1, g1, b1, s1c1);

  gemm2_k<<<dim3(BTOT / 64, 4), 128, 0, stream>>>(h1, s1c1, W2h, h2, stats2);
  bn2_fold_k<<<1, 128, 0, stream>>>(stats2, g2, b2, s2c2);

  final_k<<<(NROW * 4) / 256, 256, 0, stream>>>(h2, s2c2, out);
}
